// PointNetSetAbstractionMsg_15238543966694
// MI455X (gfx1250) — matrix operand order repaired and checked by reading, not compiled, not run on hardware
//
#include <hip/hip_runtime.h>
#include <hip/hip_bf16.h>

// ---------------------------------------------------------------------------
// PointNet++ Set Abstraction (MSG) for MI455X / gfx1250 (wave32, WMMA f16)
//
//   fps_kernel           : 16 blocks, serial 1024-step FPS, ds_swizzle argmax
//   group_kernel (x3)    : ball query + grouping -> f16 (N, 16) activation
//   per layer:
//     gemm_bn_kernel<..> : W(f16 LDS) x X(f16) via v_wmma_f32_16x16x32_f16,
//                          fused prev-layer BN+ReLU on B-fragment load,
//                          bias add, f16 store, per-channel stats via
//                          conflict-free ds_add_f32 into padded LDS arrays
//     compute_bn         : stats -> affine (a, d); zeroes stats for next layer
//   finalize_kernel (x3) : relu(max_k(a*x+d)) -> d_out feature block
// ---------------------------------------------------------------------------

typedef _Float16 half_t;
typedef __attribute__((ext_vector_type(16))) _Float16 v16h;
typedef __attribute__((ext_vector_type(8)))  _Float16 v8h;
typedef __attribute__((ext_vector_type(8)))  float    v8f;

#define NB  16
#define NPT 4096
#define NS  1024

union H16 { v16h v; _Float16 e[16]; };
union H8  { v8h  v; _Float16 e[8];  };

// xor-butterfly lane swap via ds_swizzle (group-of-32: xor=m, and=0x1f)
#define SWZ_I(x, m) __builtin_amdgcn_ds_swizzle((x), (((m) << 10) | 0x1f))

__device__ inline void argmax_step_f(float& v, int& i, float ov, int oi) {
  if (ov > v || (ov == v && oi < i)) { v = ov; i = oi; }
}

__device__ inline void argmax_reduce32(float& v, int& i) {
  { float ov = __int_as_float(SWZ_I(__float_as_int(v), 16)); int oi = SWZ_I(i, 16); argmax_step_f(v, i, ov, oi); }
  { float ov = __int_as_float(SWZ_I(__float_as_int(v), 8));  int oi = SWZ_I(i, 8);  argmax_step_f(v, i, ov, oi); }
  { float ov = __int_as_float(SWZ_I(__float_as_int(v), 4));  int oi = SWZ_I(i, 4);  argmax_step_f(v, i, ov, oi); }
  { float ov = __int_as_float(SWZ_I(__float_as_int(v), 2));  int oi = SWZ_I(i, 2);  argmax_step_f(v, i, ov, oi); }
  { float ov = __int_as_float(SWZ_I(__float_as_int(v), 1));  int oi = SWZ_I(i, 1);  argmax_step_f(v, i, ov, oi); }
}

// ---------------------------------------------------------------------------
// Farthest point sampling: one workgroup per batch, 512 threads, 8 pts/thread
// ---------------------------------------------------------------------------
__global__ __launch_bounds__(512) void fps_kernel(
    const float* __restrict__ xyz, float* __restrict__ out_newxyz,
    float* __restrict__ nx) {
  const int b = blockIdx.x;
  const int tid = threadIdx.x;
  __shared__ float lx[NPT], ly[NPT], lz[NPT];
  __shared__ float rv[16];
  __shared__ int   ri[16];
  __shared__ int   s_far;

  const float* base = xyz + (size_t)b * 3 * NPT;
  for (int i = tid; i < NPT; i += 512) {
    lx[i] = base[i]; ly[i] = base[NPT + i]; lz[i] = base[2 * NPT + i];
  }
  __syncthreads();

  // initial farthest point = argmax over x coordinate
  float bv = -1e30f; int bi = 0;
#pragma unroll
  for (int i = 0; i < 8; i++) {
    int n = tid + i * 512;
    float v = lx[n];
    if (v > bv) { bv = v; bi = n; }
  }
  argmax_reduce32(bv, bi);
  if ((tid & 31) == 0) { rv[tid >> 5] = bv; ri[tid >> 5] = bi; }
  __syncthreads();
  if (tid == 0) {
    float mv = rv[0]; int mi = ri[0];
    for (int i = 1; i < 16; i++)
      if (rv[i] > mv || (rv[i] == mv && ri[i] < mi)) { mv = rv[i]; mi = ri[i]; }
    s_far = mi;
  }
  __syncthreads();

  float dist[8];
#pragma unroll
  for (int i = 0; i < 8; i++) dist[i] = 1e10f;

  for (int s = 0; s < NS; s++) {
    const int far = s_far;
    const float cx = lx[far], cy = ly[far], cz = lz[far];
    if (tid == 0) {
      out_newxyz[(size_t)b * 3 * NS + s]          = cx;
      out_newxyz[(size_t)b * 3 * NS + NS + s]     = cy;
      out_newxyz[(size_t)b * 3 * NS + 2 * NS + s] = cz;
      float* p = nx + ((size_t)b * NS + s) * 3;
      p[0] = cx; p[1] = cy; p[2] = cz;
    }
    float bv2 = -1e30f; int bi2 = 0;
#pragma unroll
    for (int i = 0; i < 8; i++) {
      int n = tid + i * 512;
      float dx = lx[n] - cx, dy = ly[n] - cy, dz = lz[n] - cz;
      float d = dx * dx + dy * dy + dz * dz;
      float dd = fminf(dist[i], d);
      dist[i] = dd;
      if (dd > bv2) { bv2 = dd; bi2 = n; }
    }
    argmax_reduce32(bv2, bi2);
    if ((tid & 31) == 0) { rv[tid >> 5] = bv2; ri[tid >> 5] = bi2; }
    __syncthreads();
    if (tid == 0) {
      float mv = rv[0]; int mi = ri[0];
      for (int i = 1; i < 16; i++)
        if (rv[i] > mv || (rv[i] == mv && ri[i] < mi)) { mv = rv[i]; mi = ri[i]; }
      s_far = mi;
    }
    __syncthreads();
  }
}

// ---------------------------------------------------------------------------
// Ball query + grouping: one thread per center; xyz tile in LDS.
// Output: (b*NS+s)*K + k rows of 16 f16 channels:
//   [0..2]=gathered point feats, [3..5]=xyz - center, [6..15]=0 (K padding)
// ---------------------------------------------------------------------------
__global__ __launch_bounds__(256) void group_kernel(
    const float* __restrict__ xyz, const float* __restrict__ pts,
    const float* __restrict__ nx, half_t* __restrict__ out, int K, float r2) {
  const int b = blockIdx.x >> 2;
  const int s = ((blockIdx.x & 3) << 8) + threadIdx.x;
  __shared__ float lx[NPT], ly[NPT], lz[NPT];
  const float* base = xyz + (size_t)b * 3 * NPT;
  for (int i = threadIdx.x; i < NPT; i += 256) {
    lx[i] = base[i]; ly[i] = base[NPT + i]; lz[i] = base[2 * NPT + i];
  }
  __syncthreads();

  const float* c = nx + ((size_t)b * NS + s) * 3;
  const float cx = c[0], cy = c[1], cz = c[2];
  const float* pb = pts + (size_t)b * 3 * NPT;
  half_t* ob = out + ((size_t)b * NS + s) * (size_t)K * 16;

  H16 first;
#pragma unroll
  for (int i = 0; i < 16; i++) first.e[i] = (half_t)0.f;
  int cnt = 0;
  for (int n = 0; n < NPT; n++) {
    float dx = lx[n] - cx, dy = ly[n] - cy, dz = lz[n] - cz;
    float d = dx * dx + dy * dy + dz * dz;
    if (d <= r2) {
      H16 e;
      e.e[0] = (half_t)pb[n];
      e.e[1] = (half_t)pb[NPT + n];
      e.e[2] = (half_t)pb[2 * NPT + n];
      e.e[3] = (half_t)dx; e.e[4] = (half_t)dy; e.e[5] = (half_t)dz;
#pragma unroll
      for (int i = 6; i < 16; i++) e.e[i] = (half_t)0.f;
      *(v16h*)(ob + (size_t)cnt * 16) = e.v;
      if (cnt == 0) first = e;
      cnt++;
      if (cnt >= K) break;
    }
  }
  for (int k = cnt; k < K; k++) *(v16h*)(ob + (size_t)k * 16) = first.v;
}

// ---------------------------------------------------------------------------
// Stats helpers
// ---------------------------------------------------------------------------
__global__ void zero_stats(float* gsum, float* gsq) {
  int i = threadIdx.x;
  if (i < 128) { gsum[i] = 0.f; gsq[i] = 0.f; }
}

// stats -> per-channel affine (a, d); then clears stats for the next layer
__global__ void compute_bn(float* gsum, float* gsq, const float* gamma,
                           const float* beta, float* bn_a, float* bn_d,
                           float inv_cnt, int Cout) {
  int c = threadIdx.x;
  if (c < Cout) {
    float m = gsum[c] * inv_cnt;
    float v = gsq[c] * inv_cnt - m * m;
    float a = gamma[c] * rsqrtf(v + 1e-5f);
    bn_a[c] = a;
    bn_d[c] = beta[c] - m * a;
  }
  if (c < 128) { gsum[c] = 0.f; gsq[c] = 0.f; }
}

// ---------------------------------------------------------------------------
// GEMM: Y(Cout x N) = W(Cout x Cin) * act(X)  via v_wmma_f32_16x16x32_f16.
// X layout (N, CinPad) row-major f16. Per wave: one 16-col N tile, all M
// tiles. FUSE applies relu(a*x+d) of the previous layer's BN on B-load.
// Stats: ds_add_f32 into padded [row][col] LDS arrays (stride 20 f32 keeps
// the two lane halves 32 banks apart -> conflict-free), folded once per block.
// ---------------------------------------------------------------------------
#define ROWP 20

template <int MT, int KT, bool FUSE>
__global__ __launch_bounds__(256) void gemm_bn_kernel(
    const half_t* __restrict__ X, const float* __restrict__ W,
    const float* __restrict__ bias, const float* __restrict__ bn_a,
    const float* __restrict__ bn_d, half_t* __restrict__ Y,
    float* __restrict__ gsum, float* __restrict__ gsq, int ntiles, int CinReal,
    int CinPad) {
  constexpr int Cout = MT * 16;
  constexpr int Kpad = KT * 32;
  __shared__ half_t lw[128 * 96];
  __shared__ float la[96], ld[96], lb[128];
  __shared__ float lsum2[128 * ROWP];
  __shared__ float lsq2[128 * ROWP];
  const int tid = threadIdx.x;

  for (int i = tid; i < Cout * Kpad; i += 256) {
    int o = i / Kpad, c = i % Kpad;
    lw[i] = (c < CinReal) ? (half_t)W[o * CinReal + c] : (half_t)0.f;
  }
  for (int i = tid; i < Cout; i += 256) lb[i] = bias[i];
  for (int i = tid; i < Cout * ROWP; i += 256) { lsum2[i] = 0.f; lsq2[i] = 0.f; }
  if (FUSE)
    for (int i = tid; i < CinPad; i += 256) { la[i] = bn_a[i]; ld[i] = bn_d[i]; }
  __syncthreads();

  const int w = tid >> 5, lane = tid & 31, col = lane & 15, hf = lane >> 4;
  const int ntile = blockIdx.x * 8 + w;
  if (ntile < ntiles) {
    const size_t n = (size_t)ntile * 16 + col;
    const half_t* xrow = X + n * (size_t)CinPad;
    v8f acc[MT];
#pragma unroll
    for (int m = 0; m < MT; m++)
#pragma unroll
      for (int r = 0; r < 8; r++) acc[m][r] = 0.f;

#pragma unroll
    for (int kt = 0; kt < KT; kt++) {
      const int kbase = kt * 32 + hf * 16;
      H16 bf;
      if (kbase < CinPad) {
        bf.v = *(const v16h*)(xrow + kbase);
        if (FUSE) {
#pragma unroll
          for (int e = 0; e < 16; e++) {
            int c = kbase + e;
            float t = (float)bf.e[e];
            t = fmaxf(fmaf(t, la[c], ld[c]), 0.f);
            bf.e[e] = (half_t)t;
          }
        }
      } else {
#pragma unroll
        for (int e = 0; e < 16; e++) bf.e[e] = (half_t)0.f;
      }
#pragma unroll
      for (int m = 0; m < MT; m++) {
        const half_t* wr = lw + (m * 16 + col) * Kpad + kt * 32 + hf * 16;
        H8 alo, ahi;
        alo.v = *(const v8h*)(wr);
        ahi.v = *(const v8h*)(wr + 8);
        H16 af;
#pragma unroll
        for (int e = 0; e < 8; e++) { af.e[e] = alo.e[e]; af.e[8 + e] = ahi.e[e]; }
        acc[m] = __builtin_amdgcn_wmma_f32_16x16x32_f16(
            false, af.v, false, bf.v, (short)0, acc[m], false, false);
      }
    }

    // epilogue: bias, f16 store, per-channel stats into padded LDS
#pragma unroll
    for (int m = 0; m < MT; m++) {
      const int rbase = m * 16 + hf * 8;
      H8 ov;
      float vv[8];
#pragma unroll
      for (int r = 0; r < 8; r++) {
        float v = acc[m][r] + lb[rbase + r];
        vv[r] = v;
        ov.e[r] = (half_t)v;
      }
      *(v8h*)(Y + n * (size_t)Cout + rbase) = ov.v;
#pragma unroll
      for (int r = 0; r < 8; r++) {
        const int idx = (rbase + r) * ROWP + col;
        atomicAdd(&lsum2[idx], vv[r]);
        atomicAdd(&lsq2[idx], vv[r] * vv[r]);
      }
    }
  }
  __syncthreads();
  for (int i = tid; i < Cout; i += 256) {
    float s = 0.f, q = 0.f;
#pragma unroll
    for (int c2 = 0; c2 < 16; c2++) {
      s += lsum2[i * ROWP + c2];
      q += lsq2[i * ROWP + c2];
    }
    atomicAdd(&gsum[i], s);
    atomicAdd(&gsq[i], q);
  }
}

// ---------------------------------------------------------------------------
// relu(max_k(a*x+d)) -> d_out feature block (B, 320, NS)
// ---------------------------------------------------------------------------
__global__ __launch_bounds__(256) void finalize_kernel(
    const half_t* __restrict__ X, const float* __restrict__ bn_a,
    const float* __restrict__ bn_d, float* __restrict__ out, int K, int Cout,
    int choff) {
  const int idx = blockIdx.x * 256 + threadIdx.x;
  const int total = NB * NS * Cout;
  if (idx >= total) return;
  const int c = idx % Cout;
  const int s = (idx / Cout) % NS;
  const int b = idx / (Cout * NS);
  const half_t* xp = X + ((size_t)(b * NS + s) * (size_t)K) * Cout + c;
  const float a = bn_a[c], d = bn_d[c];
  float m = -1e30f;
  for (int k = 0; k < K; k++) {
    __builtin_prefetch(xp + (size_t)(k + 8) * Cout, 0, 1);  // global_prefetch_b8
    float v = fmaf((float)xp[(size_t)k * Cout], a, d);
    m = fmaxf(m, v);
  }
  out[((size_t)b * 320 + choff + c) * NS + s] = fmaxf(m, 0.f);
}

// ---------------------------------------------------------------------------
extern "C" void kernel_launch(void* const* d_in, const int* in_sizes, int n_in,
                              void* d_out, int out_size, void* d_ws,
                              size_t ws_size, hipStream_t stream) {
  (void)in_sizes; (void)n_in; (void)out_size; (void)ws_size;
  const float* xyz = (const float*)d_in[0];
  const float* pts = (const float*)d_in[1];
  auto P = [&](int sc, int ly, int it) -> const float* {
    return (const float*)d_in[2 + (sc * 3 + ly) * 4 + it];
  };
  float* out = (float*)d_out;
  char* ws = (char*)d_ws;
  float* gsum = (float*)(ws);
  float* gsq  = (float*)(ws + 512);
  float* bna  = (float*)(ws + 1024);
  float* bnd  = (float*)(ws + 1536);
  float* nx   = (float*)(ws + 4096);                 // 16*1024*3 f32
  half_t* buf0 = (half_t*)(ws + (1ull << 20));       // up to 64 MB (N x 16)
  half_t* bufA = (half_t*)(ws + (65ull << 20));      // up to 512 MB (N x 128)
  half_t* bufB = (half_t*)(ws + (577ull << 20));     // up to 384 MB (N x 96)
  float* feat = out + (size_t)NB * 3 * NS;

  fps_kernel<<<NB, 512, 0, stream>>>(xyz, out, nx);
  zero_stats<<<1, 128, 0, stream>>>(gsum, gsq);

  // ----- scale 0: r=0.1, K=16, MLP 6->32->32->64 -----
  {
    const int K = 16, Ntot = NB * NS * K, nt = Ntot / 16;
    const float inv = 1.f / (float)Ntot;
    group_kernel<<<64, 256, 0, stream>>>(xyz, pts, nx, buf0, K, 0.01f);
    gemm_bn_kernel<2, 1, false><<<(nt + 7) / 8, 256, 0, stream>>>(
        buf0, P(0, 0, 0), P(0, 0, 1), bna, bnd, bufA, gsum, gsq, nt, 6, 16);
    compute_bn<<<1, 128, 0, stream>>>(gsum, gsq, P(0, 0, 2), P(0, 0, 3), bna, bnd, inv, 32);
    gemm_bn_kernel<2, 1, true><<<(nt + 7) / 8, 256, 0, stream>>>(
        bufA, P(0, 1, 0), P(0, 1, 1), bna, bnd, bufB, gsum, gsq, nt, 32, 32);
    compute_bn<<<1, 128, 0, stream>>>(gsum, gsq, P(0, 1, 2), P(0, 1, 3), bna, bnd, inv, 32);
    gemm_bn_kernel<4, 1, true><<<(nt + 7) / 8, 256, 0, stream>>>(
        bufB, P(0, 2, 0), P(0, 2, 1), bna, bnd, bufA, gsum, gsq, nt, 32, 32);
    compute_bn<<<1, 128, 0, stream>>>(gsum, gsq, P(0, 2, 2), P(0, 2, 3), bna, bnd, inv, 64);
    const int tot = NB * NS * 64;
    finalize_kernel<<<(tot + 255) / 256, 256, 0, stream>>>(bufA, bna, bnd, feat, K, 64, 0);
  }

  // ----- scale 1: r=0.2, K=32, MLP 6->64->64->128 -----
  {
    const int K = 32, Ntot = NB * NS * K, nt = Ntot / 16;
    const float inv = 1.f / (float)Ntot;
    group_kernel<<<64, 256, 0, stream>>>(xyz, pts, nx, buf0, K, 0.04f);
    gemm_bn_kernel<4, 1, false><<<(nt + 7) / 8, 256, 0, stream>>>(
        buf0, P(1, 0, 0), P(1, 0, 1), bna, bnd, bufA, gsum, gsq, nt, 6, 16);
    compute_bn<<<1, 128, 0, stream>>>(gsum, gsq, P(1, 0, 2), P(1, 0, 3), bna, bnd, inv, 64);
    gemm_bn_kernel<4, 2, true><<<(nt + 7) / 8, 256, 0, stream>>>(
        bufA, P(1, 1, 0), P(1, 1, 1), bna, bnd, bufB, gsum, gsq, nt, 64, 64);
    compute_bn<<<1, 128, 0, stream>>>(gsum, gsq, P(1, 1, 2), P(1, 1, 3), bna, bnd, inv, 64);
    gemm_bn_kernel<8, 2, true><<<(nt + 7) / 8, 256, 0, stream>>>(
        bufB, P(1, 2, 0), P(1, 2, 1), bna, bnd, bufA, gsum, gsq, nt, 64, 64);
    compute_bn<<<1, 128, 0, stream>>>(gsum, gsq, P(1, 2, 2), P(1, 2, 3), bna, bnd, inv, 128);
    const int tot = NB * NS * 128;
    finalize_kernel<<<(tot + 255) / 256, 256, 0, stream>>>(bufA, bna, bnd, feat, K, 128, 64);
  }

  // ----- scale 2: r=0.4, K=128, MLP 6->64->96->128 -----
  {
    const int K = 128, Ntot = NB * NS * K, nt = Ntot / 16;
    const float inv = 1.f / (float)Ntot;
    group_kernel<<<64, 256, 0, stream>>>(xyz, pts, nx, buf0, K, 0.16f);
    gemm_bn_kernel<4, 1, false><<<(nt + 7) / 8, 256, 0, stream>>>(
        buf0, P(2, 0, 0), P(2, 0, 1), bna, bnd, bufA, gsum, gsq, nt, 6, 16);
    compute_bn<<<1, 128, 0, stream>>>(gsum, gsq, P(2, 0, 2), P(2, 0, 3), bna, bnd, inv, 64);
    gemm_bn_kernel<6, 2, true><<<(nt + 7) / 8, 256, 0, stream>>>(
        bufA, P(2, 1, 0), P(2, 1, 1), bna, bnd, bufB, gsum, gsq, nt, 64, 64);
    compute_bn<<<1, 128, 0, stream>>>(gsum, gsq, P(2, 1, 2), P(2, 1, 3), bna, bnd, inv, 96);
    gemm_bn_kernel<8, 3, true><<<(nt + 7) / 8, 256, 0, stream>>>(
        bufB, P(2, 2, 0), P(2, 2, 1), bna, bnd, bufA, gsum, gsq, nt, 96, 96);
    compute_bn<<<1, 128, 0, stream>>>(gsum, gsq, P(2, 2, 2), P(2, 2, 3), bna, bnd, inv, 128);
    const int tot = NB * NS * 128;
    finalize_kernel<<<(tot + 255) / 256, 256, 0, stream>>>(bufA, bna, bnd, feat, K, 128, 192);
  }
}
